// Generator_91311004713293
// MI455X (gfx1250) — matrix operand order repaired and checked by reading, not compiled, not run on hardware
//
#include <hip/hip_runtime.h>
#include <math.h>

typedef __attribute__((ext_vector_type(16))) __bf16          v16bf;
typedef __attribute__((ext_vector_type(8)))  float           v8f;
typedef int v4i __attribute__((vector_size(16)));
using u16 = unsigned short;

#define DEV __device__ __forceinline__

#if defined(__AMDGCN__) && __has_builtin(__builtin_amdgcn_global_load_async_to_lds_b128)
#define ASYNC_LDS 1
#else
#define ASYNC_LDS 0
#endif
#define AS_GLOBAL __attribute__((address_space(1)))
#define AS_LDS    __attribute__((address_space(3)))

// 16-byte global -> LDS copy: async DMA (ASYNCcnt) on CDNA5 toolchains that
// expose the builtin, else synchronous VGPR round-trip.
DEV void cp16(u16* lds, const u16* g) {
#if ASYNC_LDS
  __builtin_amdgcn_global_load_async_to_lds_b128(
      (AS_GLOBAL v4i*)g, (AS_LDS v4i*)lds, 0, 0);
#else
  *(uint4*)lds = *(const uint4*)g;
#endif
}
DEV void wait_async() {
#if ASYNC_LDS
#if __has_builtin(__builtin_amdgcn_s_wait_asynccnt)
  __builtin_amdgcn_s_wait_asynccnt(0);
#else
  asm volatile("s_wait_asynccnt 0" ::: "memory");
#endif
#endif
}

// ---------------------------------------------------------------- bf16 utils
DEV u16 f2bf(float f) {
  unsigned u = __builtin_bit_cast(unsigned, f);
  u += 0x7FFFu + ((u >> 16) & 1u);            // round-to-nearest-even
  return (u16)(u >> 16);
}
DEV float bf2f(u16 h) {
  unsigned u = ((unsigned)h) << 16;
  return __builtin_bit_cast(float, u);
}
DEV float sigmoidf_(float x) { return 1.0f / (1.0f + __expf(-x)); }

// ------------------------------------------------- one 16x16x32 bf16 WMMA tile
// A: 16x32 tile, row-major (lda in u16).  B: 32x16 tile of an [N,K] transposed
// weight (ldbt = K stride).  Per ISA 7.12.2 wave32 layouts both fragments are
// two contiguous 16B chunks per lane -> ds_load_b128 / global_load_b128:
//   A lane: row = lane&15; halfs [side*8..+8) and [16+side*8..+8)
//   B lane: col = lane&15; halfs [side*8..+8) and [16+side*8..+8)
DEV v8f wmma_tile(const u16* A, int lda, const u16* Bt, int ldbt, v8f acc) {
  const int lane = threadIdx.x & 31;
  const int rc   = lane & 15;
  const int side = lane >> 4;
  struct P { uint4 lo, hi; };
  P pa, pb;
  pa.lo = *(const uint4*)&A[(size_t)rc * lda + side * 8];
  pa.hi = *(const uint4*)&A[(size_t)rc * lda + 16 + side * 8];
  pb.lo = *(const uint4*)&Bt[(size_t)rc * ldbt + side * 8];
  pb.hi = *(const uint4*)&Bt[(size_t)rc * ldbt + 16 + side * 8];
  v16bf a = __builtin_bit_cast(v16bf, pa);
  v16bf b = __builtin_bit_cast(v16bf, pb);
  return __builtin_amdgcn_wmma_f32_16x16x32_bf16(false, a, false, b,
                                                 (short)0, acc, false, false);
}

// --------------------------------------------- 64x64 block GEMM (128 threads)
// Double-buffered K pipeline: stage k-block kb+1 into buf^1 (async global->LDS
// where available) while WMMAs consume buf; s_wait_asynccnt + barrier per step.
constexpr int LDS_ = 40;          // padded LDS stride (u16)
constexpr int BUFE = 64 * LDS_;   // one A or B buffer (u16 elems)

DEV void stage_block(u16* As, u16* Bs, const u16* __restrict__ A, int lda,
                     const u16* __restrict__ Bt, int ldbt, int k0, int tid) {
#pragma unroll
  for (int it = 0; it < 2; ++it) {  // 256 segs of 8 u16 (16B) for A and for B
    int s = tid + it * 128;
    int r = s >> 2, seg = s & 3;
    cp16(&As[r * LDS_ + seg * 8], &A[(size_t)r * lda + k0 + seg * 8]);
    cp16(&Bs[r * LDS_ + seg * 8], &Bt[(size_t)r * ldbt + k0 + seg * 8]);
  }
}

template <bool RELU, bool OUTF32>
DEV void block_gemm64x64(u16* As, u16* Bs,            // each 2*BUFE u16
                         const u16* __restrict__ A, int lda,
                         const u16* __restrict__ Bt, int ldbt,
                         const float* b1, const float* b2,
                         float* Cf, u16* Cb, int ldc, int K) {
  const int tid  = threadIdx.x;          // 128
  const int wave = tid >> 5;
  const int lane = tid & 31;
  const int side = lane >> 4;
  v8f acc[4] = {};
  const int nk = K >> 5;
  stage_block(As, Bs, A, lda, Bt, ldbt, 0, tid);
  wait_async();
  __syncthreads();
  int buf = 0;
  for (int kb = 0; kb < nk; ++kb) {
    if (kb + 1 < nk)
      stage_block(As + (buf ^ 1) * BUFE, Bs + (buf ^ 1) * BUFE,
                  A, lda, Bt, ldbt, (kb + 1) * 32, tid);
    const u16* a  = As + buf * BUFE + (wave * 16) * LDS_;
    const u16* bb = Bs + buf * BUFE;
#pragma unroll
    for (int j = 0; j < 4; ++j)
      acc[j] = wmma_tile(a, LDS_, bb + (j * 16) * LDS_, LDS_, acc[j]);
    wait_async();
    __syncthreads();
    buf ^= 1;
  }
#pragma unroll
  for (int j = 0; j < 4; ++j) {
#pragma unroll
    for (int r = 0; r < 8; ++r) {
      int m = wave * 16 + r + side * 8;  // C/D layout: VGPR r -> M=r / r+8
      int n = j * 16 + (lane & 15);
      float v = acc[j][r];
      if (b1) v += b1[n];
      if (b2) v += b2[n];
      if (RELU) v = v > 0.f ? v : 0.f;
      if (OUTF32) Cf[(size_t)m * ldc + n] = v;
      else        Cb[(size_t)m * ldc + n] = f2bf(v);
    }
  }
}

template <bool RELU, bool OUTF32>
__global__ void gemm_kernel(const u16* __restrict__ A, int lda,
                            const u16* __restrict__ Bt, int ldbt,
                            const float* __restrict__ b1,
                            const float* __restrict__ b2,
                            float* __restrict__ Cf, u16* __restrict__ Cb,
                            int ldc, int K) {
  __shared__ __align__(16) u16 As[2 * BUFE];
  __shared__ __align__(16) u16 Bs[2 * BUFE];
  int mb = blockIdx.x, nb = blockIdx.y;
  block_gemm64x64<RELU, OUTF32>(
      As, Bs, A + (size_t)mb * 64 * lda, lda,
      Bt + (size_t)nb * 64 * ldbt, ldbt,
      b1 ? b1 + nb * 64 : nullptr, b2 ? b2 + nb * 64 : nullptr,
      OUTF32 ? Cf + (size_t)mb * 64 * ldc + nb * 64 : nullptr,
      OUTF32 ? nullptr : Cb + (size_t)mb * 64 * ldc + nb * 64, ldc, K);
}

// ----------------------------------- per-sample expert GEMM (argmax(label))
__global__ void expert_kernel(const u16* __restrict__ combined,  // [R,576]
                              const u16* __restrict__ WcT,       // [16,512,576]
                              const float* __restrict__ bc,      // [16,512]
                              const float* __restrict__ label,   // [B,16]
                              u16* __restrict__ out) {           // [R,512]
  __shared__ __align__(16) u16 As[2 * BUFE];
  __shared__ __align__(16) u16 Bs[2 * BUFE];
  __shared__ int lab;
  const int b = blockIdx.x, nb = blockIdx.y;
  if (threadIdx.x == 0) {
    const float* L = label + b * 16;
    int best = 0; float bv = L[0];
    for (int i = 1; i < 16; ++i) if (L[i] > bv) { bv = L[i]; best = i; }
    lab = best;
  }
  __syncthreads();
  const int l = lab;
  block_gemm64x64<true, false>(
      As, Bs, combined + (size_t)b * 64 * 576, 576,
      WcT + (size_t)l * 512 * 576 + (size_t)nb * 64 * 576, 576,
      bc + l * 512 + nb * 64, nullptr, nullptr,
      out + (size_t)b * 64 * 512 + nb * 64, 512, 576);
}

// --------------------------------------------------- persistent LSTM scan
// One block = 16 batch rows, 8 waves (256 thr). LDS (dynamic, 176KB -- needs
// CDNA5's 320KB WGP LDS): gate buffer 16x2048 f32, c 16x512 f32, h 16x512 bf16.
// Whh is pre-transposed [2048,512] so B tiles are contiguous b128 global loads
// (L2-resident, 192MB L2), with global_prefetch ahead of the K stream.
__global__ void lstm_scan_kernel(const u16* __restrict__ pre,    // [R,2048] bf16
                                 const u16* __restrict__ WhhT,   // [2048,512] bf16
                                 u16* __restrict__ hseq,         // [R,512] bf16
                                 int T) {
  extern __shared__ char smem[];
  float* gbuf = (float*)smem;                              // 16*2048*4
  float* cst  = (float*)(smem + 16 * 2048 * 4);            // 16*512*4
  u16*   hst  = (u16*)(smem + 16 * 2048 * 4 + 16 * 512 * 4);
  const int tid  = threadIdx.x;  // 256
  const int wave = tid >> 5;
  const int lane = tid & 31;
  const int side = lane >> 4;
  const int b0   = blockIdx.x * 16;
  for (int i = tid; i < 16 * 512; i += 256) { cst[i] = 0.f; hst[i] = 0; }
  __syncthreads();
  for (int t = 0; t < T; ++t) {
    const int cw = wave * 256;
    for (int j = 0; j < 16; ++j) {
      const int cb = cw + j * 16;
      const u16* Bcol = WhhT + (size_t)cb * 512;   // [16 cols][512 k]
      v8f acc = {};
      for (int kb = 0; kb < 16; ++kb) {
        __builtin_prefetch(Bcol + (((kb + 1) * 32) & 511), 0, 0);
        acc = wmma_tile(hst, 512, Bcol + kb * 32, 512, acc);
      }
#pragma unroll
      for (int r = 0; r < 8; ++r) {
        int m = r + side * 8;
        int col = cb + (lane & 15);
        float g = acc[r] + bf2f(pre[((size_t)(b0 + m) * T + t) * 2048 + col]);
        gbuf[m * 2048 + col] = g;
      }
    }
    __syncthreads();
    for (int i = tid; i < 16 * 512; i += 256) {
      int m = i >> 9, hc = i & 511;
      float gi = gbuf[m * 2048 + hc];
      float gf = gbuf[m * 2048 + 512 + hc];
      float gg = gbuf[m * 2048 + 1024 + hc];
      float go = gbuf[m * 2048 + 1536 + hc];
      float c = sigmoidf_(gf) * cst[i] + sigmoidf_(gi) * tanhf(gg);
      float h = sigmoidf_(go) * tanhf(c);
      cst[i] = c;
      u16 hb = f2bf(h);
      hst[i] = hb;
      hseq[((size_t)(b0 + m) * T + t) * 512 + hc] = hb;
    }
    __syncthreads();
  }
}

// ------------------------------------------------------------- small kernels
__global__ void embed_kernel(const int* __restrict__ x, const int* __restrict__ xn,
                             const int* __restrict__ length,
                             const float* e0, const float* e1, const float* e2,
                             const float* e3, const float* n0, const float* n1,
                             const float* n2, const float* n3,
                             const float* __restrict__ lenW,
                             const float* __restrict__ lenb,
                             u16* __restrict__ combined,   // [R,576]
                             u16* __restrict__ nowe) {     // [R,512]
  const int row = blockIdx.x;      // b*64 + t
  const int b   = row >> 6;
  const int tid = threadIdx.x;     // 128
  const float* E[4] = {e0, e1, e2, e3};
  const float* N[4] = {n0, n1, n2, n3};
#pragma unroll
  for (int i = 0; i < 4; ++i) {
    int id  = x[row * 4 + i];
    int idn = xn[row * 4 + i];
    combined[(size_t)row * 576 + i * 128 + tid] = f2bf(E[i][(size_t)id * 128 + tid]);
    nowe[(size_t)row * 512 + i * 128 + tid]     = f2bf(N[i][(size_t)idn * 128 + tid]);
  }
  if (tid < 64) {
    int lr = length[b] - 1;
    float v = lenW[lr * 64 + tid] + lenb[tid];
    combined[(size_t)row * 576 + 512 + tid] = f2bf(v > 0.f ? v : 0.f);
  }
}

// fp32 [K,N] row-major  ->  bf16 [N,K] (transposed, column-major weight)
__global__ void cvt_t_kernel(const float* __restrict__ s, u16* __restrict__ d,
                             int K, int N) {
  int total = K * N;
  for (int i = blockIdx.x * blockDim.x + threadIdx.x; i < total;
       i += gridDim.x * blockDim.x) {
    int k = i / N, n = i - k * N;
    d[(size_t)n * K + k] = f2bf(s[i]);
  }
}
__global__ void zero_u16_kernel(u16* __restrict__ d, int n) {
  for (int i = blockIdx.x * blockDim.x + threadIdx.x; i < n;
       i += gridDim.x * blockDim.x)
    d[i] = 0;
}
__global__ void copy_cols_kernel(const u16* __restrict__ s, int lds_,
                                 u16* __restrict__ d, int ldd, int doff,
                                 int ncols, int total) {
  for (int i = blockIdx.x * blockDim.x + threadIdx.x; i < total;
       i += gridDim.x * blockDim.x) {
    int r = i / ncols, c = i - r * ncols;
    d[(size_t)r * ldd + doff + c] = s[(size_t)r * lds_ + c];
  }
}
__global__ void softmax_kernel(float* __restrict__ p, int ldc, int v) {
  float* q = p + (size_t)blockIdx.x * ldc;
  __shared__ float red[128];
  const int tid = threadIdx.x;
  float m = -3.4e38f;
  for (int c = tid; c < v; c += 128) m = fmaxf(m, q[c]);
  red[tid] = m; __syncthreads();
  for (int s = 64; s > 0; s >>= 1) {
    if (tid < s) red[tid] = fmaxf(red[tid], red[tid + s]);
    __syncthreads();
  }
  m = red[0]; __syncthreads();
  float sum = 0.f;
  for (int c = tid; c < v; c += 128) { float e = __expf(q[c] - m); q[c] = e; sum += e; }
  red[tid] = sum; __syncthreads();
  for (int s = 64; s > 0; s >>= 1) {
    if (tid < s) red[tid] += red[tid + s];
    __syncthreads();
  }
  float inv = 1.f / red[0];
  for (int c = tid; c < v; c += 128) q[c] *= inv;
}

// ------------------------------------------------------------- workspace map
constexpr size_t RR = 256 * 64;  // 16384 rows (B*T)
constexpr size_t al(size_t x) { return (x + 255) & ~size_t(255); }
constexpr size_t OFF_WCOMB = 0;                                         // 16*512*576 bf16 (T)
constexpr size_t OFF_WIH   = al(OFF_WCOMB + size_t(16) * 576 * 512 * 2);
constexpr size_t OFF_WHH   = al(OFF_WIH + size_t(4) * 512 * 2048 * 2);
constexpr size_t OFF_WOUT  = al(OFF_WHH + size_t(4) * 512 * 2048 * 2);
constexpr size_t OFF_WCOND = al(OFF_WOUT + size_t(576) * 1600 * 2);
constexpr size_t OFF_COMB  = al(OFF_WCOND + size_t(128 + 256 + 384) * 64 * 2);
constexpr size_t OFF_NOW   = al(OFF_COMB + RR * 576 * 2);
constexpr size_t OFF_ATTR  = al(OFF_NOW + RR * 512 * 2);
constexpr size_t OFF_ACTX  = al(OFF_ATTR + 4 * RR * 64 * 2);
constexpr size_t OFF_ACTH  = al(OFF_ACTX + RR * 512 * 2);
constexpr size_t OFF_PRE   = al(OFF_ACTH + RR * 512 * 2);
constexpr int SCAN_SMEM = 16 * 2048 * 4 + 16 * 512 * 4 + 16 * 512 * 2;  // 176KB

extern "C" void kernel_launch(void* const* d_in, const int* in_sizes, int n_in,
                              void* d_out, int out_size, void* d_ws,
                              size_t ws_size, hipStream_t stream) {
  (void)in_sizes; (void)n_in; (void)out_size; (void)ws_size;
  char* ws = (char*)d_ws;
  u16* WcombT = (u16*)(ws + OFF_WCOMB);
  u16* WihT   = (u16*)(ws + OFF_WIH);
  u16* WhhT   = (u16*)(ws + OFF_WHH);
  u16* WoutT  = (u16*)(ws + OFF_WOUT);
  u16* WcondT = (u16*)(ws + OFF_WCOND);
  u16* comb   = (u16*)(ws + OFF_COMB);   // combined, later reused as cat buffer
  u16* nowe   = (u16*)(ws + OFF_NOW);
  u16* attr   = (u16*)(ws + OFF_ATTR);
  u16* actx   = (u16*)(ws + OFF_ACTX);
  u16* acth   = (u16*)(ws + OFF_ACTH);
  u16* pre    = (u16*)(ws + OFF_PRE);

  auto cvt_t = [&](const void* src, u16* dst, int K, int N) {
    int n = K * N;
    cvt_t_kernel<<<(n + 255) / 256, 256, 0, stream>>>((const float*)src, dst, K, N);
  };
  // weight conversion fp32 [K,N] -> bf16 [N,K], every call (deterministic)
  for (int l = 0; l < 16; ++l)
    cvt_t((const float*)d_in[22] + (size_t)l * 576 * 512,
          WcombT + (size_t)l * 512 * 576, 576, 512);
  for (int l = 0; l < 4; ++l)
    cvt_t(d_in[24 + l], WihT + (size_t)l * 512 * 2048, 512, 2048);
  for (int l = 0; l < 4; ++l)
    cvt_t(d_in[28 + l], WhhT + (size_t)l * 512 * 2048, 512, 2048);
  const int    NS[4] = {64, 256, 1024, 256};
  const int    CO[4] = {0, 64, 320, 1344};
  const size_t WO[4] = {0, 576 * 64, 576 * (64 + 256), 576 * (64 + 256 + 1024)};
  for (int f = 0; f < 4; ++f) cvt_t(d_in[40 + f], WoutT + WO[f], 576, NS[f]);
  const size_t CND[3] = {0, size_t(128) * 64, size_t(128 + 256) * 64};
  for (int i = 0; i < 3; ++i)
    cvt_t(d_in[12 + i], WcondT + CND[i], (i + 1) * 128, 64);

  // embeddings + length branch -> combined [R,576], now_emb [R,512]
  embed_kernel<<<(int)RR, 128, 0, stream>>>(
      (const int*)d_in[2], (const int*)d_in[3], (const int*)d_in[1],
      (const float*)d_in[4], (const float*)d_in[5], (const float*)d_in[6],
      (const float*)d_in[7], (const float*)d_in[8], (const float*)d_in[9],
      (const float*)d_in[10], (const float*)d_in[11],
      (const float*)d_in[20], (const float*)d_in[21], comb, nowe);

  // attr features: field0 = zeros, field i = relu(now_emb[:, :i*128] @ cond_W[i-1])
  zero_u16_kernel<<<4096, 256, 0, stream>>>(attr, (int)(RR * 64));
  for (int i = 1; i <= 3; ++i)
    gemm_kernel<true, false><<<dim3(RR / 64, 1), 128, 0, stream>>>(
        nowe, 512, WcondT + CND[i - 1], i * 128,
        (const float*)d_in[16 + (i - 1)], nullptr, nullptr,
        attr + (size_t)i * RR * 64, 64, i * 128);

  // per-sample expert combine: relu(combined @ comb_W[argmax(label)]) -> actx
  expert_kernel<<<dim3(256, 8), 128, 0, stream>>>(
      comb, WcombT, (const float*)d_in[23], (const float*)d_in[0], actx);

  // 4-layer LSTM
  for (int l = 0; l < 4; ++l) {
    gemm_kernel<false, false><<<dim3(RR / 64, 32), 128, 0, stream>>>(
        actx, 512, WihT + (size_t)l * 512 * 2048, 512,
        (const float*)d_in[32 + l], (const float*)d_in[36 + l], nullptr, pre,
        2048, 512);
    lstm_scan_kernel<<<16, 256, SCAN_SMEM, stream>>>(
        pre, WhhT + (size_t)l * 512 * 2048, acth, 64);
    u16* tmp = actx; actx = acth; acth = tmp;
  }
  // hidden = actx

  // output heads: cat = [hidden | attr_f] (reuse combined buffer), then GEMM+softmax
  copy_cols_kernel<<<8192, 256, 0, stream>>>(actx, 512, comb, 576, 0, 512,
                                             (int)(RR * 512));
  for (int f = 0; f < 4; ++f) {
    copy_cols_kernel<<<4096, 256, 0, stream>>>(attr + (size_t)f * RR * 64, 64,
                                               comb, 576, 512, 64,
                                               (int)(RR * 64));
    gemm_kernel<false, true><<<dim3(RR / 64, NS[f] / 64), 128, 0, stream>>>(
        comb, 576, WoutT + WO[f], 576, (const float*)d_in[44 + f], nullptr,
        (float*)d_out + CO[f], nullptr, 1600, 576);
    softmax_kernel<<<(int)RR, 128, 0, stream>>>((float*)d_out + CO[f], 1600,
                                                NS[f]);
  }
}
